// RefIndexer_2911987827138
// MI455X (gfx1250) — matrix operand order repaired and checked by reading, not compiled, not run on hardware
//
#include <hip/hip_runtime.h>
#include <hip/hip_bf16.h>
#include <stdint.h>

// ---------------- problem constants ----------------
#define S_LEN   2048
#define DIM_E   4096
#define QLORA_R 1536
#define NHEAD   32
#define DHEAD   128
#define ROPE_D  64
#define TOPK_N  512
#define LN_EPS  1e-6f
#define NEG_INF (-1000000000.0f)

typedef __bf16 bf16;
typedef __attribute__((ext_vector_type(8)))  bf16  v8bf;
typedef __attribute__((ext_vector_type(16))) bf16  v16bf;
typedef __attribute__((ext_vector_type(8)))  float v8f;
typedef __attribute__((ext_vector_type(4)))  unsigned short us4;
typedef __attribute__((ext_vector_type(4)))  unsigned int   u32x4;
typedef __attribute__((ext_vector_type(8)))  int            i32x8;
typedef __attribute__((ext_vector_type(4)))  int            i32x4;

// ---- Tensor Data Mover availability / arity detection ----
#if defined(__has_builtin)
#  if __has_builtin(__builtin_amdgcn_tensor_load_to_lds) && __has_builtin(__builtin_amdgcn_s_wait_tensorcnt)
#    define USE_TDM 1
#  endif
#endif
#ifndef USE_TDM
#  define USE_TDM 0
#endif
#if USE_TDM
#  if __has_include(<hip/amd_detail/amd_gfx1250_TDM.h>)
#    define TDM_SIX_ARGS 1   /* therock-10.0 headers -> clang-23 6-arg builtin */
#  else
#    define TDM_SIX_ARGS 0   /* ROCm 7.2 clang-22 -> 5-arg builtin */
#  endif
#endif

// ---------------- helpers ----------------
__device__ __forceinline__ unsigned short f2bfu(float f) {
    unsigned u = __builtin_bit_cast(unsigned, f);
    unsigned r = u + 0x7FFFu + ((u >> 16) & 1u);   // round-to-nearest-even
    return (unsigned short)(r >> 16);
}
__device__ __forceinline__ bf16 f2bf(float f) {
    unsigned short h = f2bfu(f);
    return __builtin_bit_cast(bf16, h);
}

// A-matrix 16x32 bf16 fragment (ISA 7.12.2): lane&15 = M row; lanes 0-15 take
// K {0..7,16..23}, lanes 16-31 take K {8..15,24..31}.
__device__ __forceinline__ v16bf load_fragA(const bf16* __restrict__ row, int lane) {
    const int half = lane >> 4;
    v8bf lo = *(const v8bf*)(row + half * 8);
    v8bf hi = *(const v8bf*)(row + 16 + half * 8);
    return __builtin_shufflevector(lo, hi, 0,1,2,3,4,5,6,7,8,9,10,11,12,13,14,15);
}
// B-matrix 32x16 bf16 fragment: lane&15 = N col; lanes 0-15 take
// K {0..7,16..23}, lanes 16-31 take K {8..15,24..31} (same as A).
__device__ __forceinline__ v16bf load_fragB(const bf16* __restrict__ row, int lane) {
    const int half = lane >> 4;
    v8bf lo = *(const v8bf*)(row + half * 8);
    v8bf hi = *(const v8bf*)(row + 16 + half * 8);
    return __builtin_shufflevector(lo, hi, 0,1,2,3,4,5,6,7,8,9,10,11,12,13,14,15);
}
__device__ __forceinline__ v8f wmma_bf16(v16bf a, v16bf b, v8f c) {
    return __builtin_amdgcn_wmma_f32_16x16x32_bf16(false, a, false, b, (short)0, c, false, false);
}

// ---------------- fp32 -> bf16 convert ----------------
__global__ void cvt_f32_bf16(const float* __restrict__ in, unsigned short* __restrict__ out, int n) {
    int i = (blockIdx.x * blockDim.x + threadIdx.x) * 4;
    if (i + 3 < n) {
        float4 v = *(const float4*)(in + i);
        us4 o = { f2bfu(v.x), f2bfu(v.y), f2bfu(v.z), f2bfu(v.w) };
        *(us4*)(out + i) = o;
    } else {
        for (; i < n; ++i) out[i] = f2bfu(in[i]);
    }
}

// ------- generic bf16 WMMA GEMM: C[M,N] = A[M,K] * B[N,K]^T -------
// MT x NT register tiling: MT+NT fragment loads feed MT*NT WMMAs per K-step.
// <2,4> -> 32x64 per wave, 12 b128 loads : 8 WMMAs (~130 VGPRs, no spill).
template<int MT, int NT>
__global__ void gemm_bf16_wmma(const bf16* __restrict__ A, const bf16* __restrict__ B,
                               float* __restrict__ C, int M, int N, int K, float scale) {
    const int lane = threadIdx.x & 31;
    const int wave = threadIdx.x >> 5;
    const int n0 = (blockIdx.x * (blockDim.x >> 5) + wave) * (16 * NT);
    const int m0 = blockIdx.y * (16 * MT);
    if (n0 >= N || m0 >= M) return;   // wave-uniform exit (EXEC stays all-ones)

    const bf16* ar[MT];
    const bf16* br[NT];
    #pragma unroll
    for (int mt = 0; mt < MT; ++mt) ar[mt] = A + (size_t)(m0 + mt * 16 + (lane & 15)) * K;
    #pragma unroll
    for (int nt = 0; nt < NT; ++nt) br[nt] = B + (size_t)(n0 + nt * 16 + (lane & 15)) * K;

    v8f acc[MT][NT] = {};
    for (int k0 = 0; k0 < K; k0 += 32) {
        v16bf a[MT], b[NT];
        #pragma unroll
        for (int mt = 0; mt < MT; ++mt) a[mt] = load_fragA(ar[mt] + k0, lane);
        #pragma unroll
        for (int nt = 0; nt < NT; ++nt) b[nt] = load_fragB(br[nt] + k0, lane);
        #pragma unroll
        for (int mt = 0; mt < MT; ++mt)
            #pragma unroll
            for (int nt = 0; nt < NT; ++nt)
                acc[mt][nt] = wmma_bf16(a[mt], b[nt], acc[mt][nt]);
    }
    // D layout per 16x16 tile: lane&15 = N col; lanes<16 -> rows +0..7, lanes>=16 -> +8..15
    const int nc = lane & 15;
    const int mb = (lane >> 4) << 3;
    #pragma unroll
    for (int mt = 0; mt < MT; ++mt)
        #pragma unroll
        for (int nt = 0; nt < NT; ++nt) {
            float* crow = C + (size_t)(m0 + mt * 16 + mb) * N + (n0 + nt * 16 + nc);
            #pragma unroll
            for (int e = 0; e < 8; ++e) crow[(size_t)e * N] = acc[mt][nt][e] * scale;
        }
}

// ---------------- k post: LayerNorm + RoPE, fp32 -> bf16 ----------------
__global__ void kpost_ln_rope(const float* __restrict__ kf,
                              const float* __restrict__ lnw, const float* __restrict__ lnb,
                              const float* __restrict__ cosT, const float* __restrict__ sinT,
                              bf16* __restrict__ kb) {
    const int lane = threadIdx.x & 31;
    const int s = blockIdx.x * (blockDim.x >> 5) + (threadIdx.x >> 5);
    if (s >= S_LEN) return;
    const float* row = kf + (size_t)s * DHEAD;
    float v0 = row[lane], v1 = row[lane + 32], v2 = row[lane + 64], v3 = row[lane + 96];
    float sum = v0 + v1 + v2 + v3;
    float sq  = v0*v0 + v1*v1 + v2*v2 + v3*v3;
    #pragma unroll
    for (int m = 16; m; m >>= 1) { sum += __shfl_xor(sum, m, 32); sq += __shfl_xor(sq, m, 32); }
    const float mu  = sum * (1.0f / DHEAD);
    const float var = sq * (1.0f / DHEAD) - mu * mu;
    const float inv = rsqrtf(var + LN_EPS);
    float n0 = (v0 - mu) * inv * lnw[lane]      + lnb[lane];
    float n1 = (v1 - mu) * inv * lnw[lane + 32] + lnb[lane + 32];
    float n2 = (v2 - mu) * inv * lnw[lane + 64] + lnb[lane + 64];
    float n3 = (v3 - mu) * inv * lnw[lane + 96] + lnb[lane + 96];
    const float c  = cosT[s * (ROPE_D / 2) + lane];
    const float sn = sinT[s * (ROPE_D / 2) + lane];
    bf16* orow = kb + (size_t)s * DHEAD;
    orow[lane]      = f2bf(n0 * c - n1 * sn);
    orow[lane + 32] = f2bf(n0 * sn + n1 * c);
    orow[lane + 64] = f2bf(n2);
    orow[lane + 96] = f2bf(n3);
}

// ---------------- q post: RoPE, fp32 -> bf16 (row r = s*NHEAD + h) ----------------
__global__ void qpost_rope(const float* __restrict__ qf,
                           const float* __restrict__ cosT, const float* __restrict__ sinT,
                           bf16* __restrict__ qb) {
    const int lane = threadIdx.x & 31;
    const int r = blockIdx.x * (blockDim.x >> 5) + (threadIdx.x >> 5);
    if (r >= S_LEN * NHEAD) return;
    const int s = r >> 5;   // NHEAD == 32
    const float* row = qf + (size_t)r * DHEAD;
    float v0 = row[lane], v1 = row[lane + 32], v2 = row[lane + 64], v3 = row[lane + 96];
    const float c  = cosT[s * (ROPE_D / 2) + lane];
    const float sn = sinT[s * (ROPE_D / 2) + lane];
    bf16* orow = qb + (size_t)r * DHEAD;
    orow[lane]      = f2bf(v0 * c - v1 * sn);
    orow[lane + 32] = f2bf(v0 * sn + v1 * c);
    orow[lane + 64] = f2bf(v2);
    orow[lane + 96] = f2bf(v3);
}

// ------- fused scores: sum_h relu(q_h . k / sqrt(DH)) * wT[h,s] + causal -------
// Block: 256 threads = 8 waves. Block covers t-tile of 32 (k staged in LDS via
// the Tensor Data Mover) and 128 s-rows (16 per wave). B fragments are
// head-invariant: preloaded once into registers, so the head loop is pure
// A-loads + WMMA (8 WMMAs per head per wave). Weights are h-major (wgtT) so a
// lane's 8 weights are one contiguous 32B run: 2 b128 loads, 1 address pair.
__global__ void score_wmma(const bf16* __restrict__ qb, const bf16* __restrict__ kb,
                           const float* __restrict__ wgtT, float* __restrict__ isc) {
    __shared__ bf16 lk[32 * DHEAD];   // 8 KB k tile (32 t-rows x 128 d)
    const int lane = threadIdx.x & 31;
    const int wave = threadIdx.x >> 5;
    const int t0 = blockIdx.x * 32;
    const int s0 = blockIdx.y * 128 + wave * 16;

#if USE_TDM
    if (wave == 0) {
        // D# for a 2D tile: element = 4 bytes; tensor 64 dwords x 2048 rows,
        // tile 64 dwords x 32 rows starting at row t0.
        const unsigned long long ga = (unsigned long long)(uintptr_t)(kb + (size_t)t0 * DHEAD);
        const unsigned lds_off = (unsigned)(uintptr_t)(void*)lk;
        u32x4 g0 = { 1u,                                  // count=1, user mode
                     lds_off,                             // lds_addr
                     (unsigned)ga,                        // global_addr[31:0]
                     ((unsigned)(ga >> 32) & 0x01FFFFFFu) | 0x80000000u }; // [56:32] | type=2
        i32x8 g1 = { (int)(2u << 16),       // data_size = 4B, no multicast/pad/iterate
                     (int)(64u << 16),      // tensor_dim0[15:0] = 64 dwords/row
                     (int)(2048u << 16),    // tensor_dim1[15:0] = 2048 rows
                     (int)(64u << 16),      // tile_dim0 = 64 dwords
                     (int)32,               // tile_dim1 = 32 rows
                     (int)64,               // tensor_dim0_stride = 64 dwords
                     0, 0 };
        i32x4 gz = { 0, 0, 0, 0 };
#if TDM_SIX_ARGS
        i32x8 gz8 = { 0, 0, 0, 0, 0, 0, 0, 0 };
        __builtin_amdgcn_tensor_load_to_lds(g0, g1, gz, gz, gz8, 0);
#else
        __builtin_amdgcn_tensor_load_to_lds(g0, g1, gz, gz, 0);
#endif
        __builtin_amdgcn_s_wait_tensorcnt(0);
    }
    __syncthreads();
#else
    {   // cooperative stage: 4096 bf16, 16 per thread
        const int idx = threadIdx.x * 16;
        const int tr = idx >> 7, tc = idx & 127;
        *(v8bf*)&lk[idx]     = *(const v8bf*)(kb + (size_t)(t0 + tr) * DHEAD + tc);
        *(v8bf*)&lk[idx + 8] = *(const v8bf*)(kb + (size_t)(t0 + tr) * DHEAD + tc + 8);
    }
    __syncthreads();
#endif

    const int mrow  = lane & 15;
    const int half8 = (lane >> 4) << 3;
    const float inv_dh = 0.08838834764831845f;   // DHEAD^-0.5

    // Preload head-invariant B fragments: 2 t-subtiles x 4 K-steps
    v16bf bf[2][4];
    #pragma unroll
    for (int tt = 0; tt < 2; ++tt) {
        const bf16* brow = &lk[((tt << 4) + (lane & 15)) * DHEAD];
        #pragma unroll
        for (int kk = 0; kk < 4; ++kk) bf[tt][kk] = load_fragB(brow + kk * 32, lane);
    }

    const float* wbase = wgtT + s0 + half8;   // 8 consecutive weights per lane per head
    float fin[2][8] = {};
    for (int h = 0; h < NHEAD; ++h) {
        const bf16* arow = qb + ((size_t)(s0 + mrow) * NHEAD + h) * DHEAD;
        v8f acc0 = {}, acc1 = {};
        #pragma unroll
        for (int kk = 0; kk < 4; ++kk) {
            v16bf a = load_fragA(arow + kk * 32, lane);
            acc0 = wmma_bf16(a, bf[0][kk], acc0);
            acc1 = wmma_bf16(a, bf[1][kk], acc1);
        }
        float wv[8];
        *(float4*)&wv[0] = *(const float4*)(wbase + (size_t)h * S_LEN);
        *(float4*)&wv[4] = *(const float4*)(wbase + (size_t)h * S_LEN + 4);
        #pragma unroll
        for (int e = 0; e < 8; ++e) {
            float s0c = acc0[e] * inv_dh; s0c = s0c > 0.f ? s0c : 0.f;
            float s1c = acc1[e] * inv_dh; s1c = s1c > 0.f ? s1c : 0.f;
            fin[0][e] += s0c * wv[e];
            fin[1][e] += s1c * wv[e];
        }
    }
    #pragma unroll
    for (int tt = 0; tt < 2; ++tt) {
        const int t = t0 + tt * 16 + (lane & 15);
        #pragma unroll
        for (int e = 0; e < 8; ++e) {
            const int s = s0 + half8 + e;
            isc[(size_t)s * S_LEN + t] = fin[tt][e] + (t <= s ? 0.f : NEG_INF);
        }
    }
}

// ---------------- top-512 per row via in-LDS bitonic sort of 2048 keys ----------------
__global__ void topk_bitonic(const float* __restrict__ isc, int* __restrict__ out) {
    __shared__ uint64_t key[S_LEN];   // 16 KB
    const int row = blockIdx.x;
    const float* r = isc + (size_t)row * S_LEN;
    for (int i = threadIdx.x; i < S_LEN; i += blockDim.x) {
        unsigned b = __builtin_bit_cast(unsigned, r[i]);
        unsigned u = (b & 0x80000000u) ? ~b : (b | 0x80000000u);  // order-preserving map
        unsigned iu = ~u;                                         // ascending key == descending value
        key[i] = ((uint64_t)iu << 32) | (unsigned)i;              // tie-break: lower index first
    }
    __syncthreads();
    for (int k = 2; k <= S_LEN; k <<= 1) {
        for (int j = k >> 1; j > 0; j >>= 1) {
            for (int i = threadIdx.x; i < S_LEN; i += blockDim.x) {
                const int ixj = i ^ j;
                if (ixj > i) {
                    uint64_t a = key[i], b2 = key[ixj];
                    const bool asc = ((i & k) == 0);
                    if ((a > b2) == asc) { key[i] = b2; key[ixj] = a; }
                }
            }
            __syncthreads();
        }
    }
    for (int i = threadIdx.x; i < TOPK_N; i += blockDim.x)
        out[(size_t)row * TOPK_N + i] = (int)(unsigned)(key[i] & 0xFFFFFFFFu);
}

// ---------------- host launcher ----------------
extern "C" void kernel_launch(void* const* d_in, const int* in_sizes, int n_in,
                              void* d_out, int out_size, void* d_ws, size_t ws_size,
                              hipStream_t stream) {
    (void)in_sizes; (void)n_in; (void)out_size; (void)ws_size;
    const float* x        = (const float*)d_in[0];   // (S, DIM)
    const float* qr       = (const float*)d_in[1];   // (S, QLORA)
    const float* wq_b     = (const float*)d_in[2];   // (H*DH, QLORA)
    const float* wk       = (const float*)d_in[3];   // (DH, DIM)
    const float* k_norm_w = (const float*)d_in[4];   // (DH,)
    const float* k_norm_b = (const float*)d_in[5];   // (DH,)
    const float* w_wts    = (const float*)d_in[6];   // (H, DIM)
    const float* cosT     = (const float*)d_in[7];   // (S, ROPE/2)
    const float* sinT     = (const float*)d_in[8];   // (S, ROPE/2)
    /* d_in[9] = mask: causal, computed analytically */

    // workspace carve-out (256B aligned)
    char* ws = (char*)d_ws;
    size_t off = 0;
    auto carve = [&](size_t bytes) -> void* {
        void* p = ws + off;
        off = (off + bytes + 255) & ~(size_t)255;
        return p;
    };
    bf16*  x_bf   = (bf16*) carve((size_t)S_LEN * DIM_E   * 2);
    bf16*  qr_bf  = (bf16*) carve((size_t)S_LEN * QLORA_R * 2);
    bf16*  wqb_bf = (bf16*) carve((size_t)NHEAD * DHEAD * QLORA_R * 2);
    bf16*  wk_bf  = (bf16*) carve((size_t)DHEAD * DIM_E * 2);
    bf16*  ww_bf  = (bf16*) carve((size_t)NHEAD * DIM_E * 2);
    float* q_f32  = (float*)carve((size_t)S_LEN * NHEAD * DHEAD * 4);
    float* k_f32  = (float*)carve((size_t)S_LEN * DHEAD * 4);
    float* wgtT   = (float*)carve((size_t)NHEAD * S_LEN * 4);   // h-major (transposed)
    bf16*  q_bf   = (bf16*) carve((size_t)S_LEN * NHEAD * DHEAD * 2);
    bf16*  k_bf   = (bf16*) carve((size_t)S_LEN * DHEAD * 2);
    float* isc    = (float*)carve((size_t)S_LEN * S_LEN * 4);

    // 1) convert GEMM operands to bf16
    auto cvt = [&](const float* src, bf16* dst, int n) {
        int blocks = (n / 4 + 255) / 256;
        cvt_f32_bf16<<<blocks, 256, 0, stream>>>(src, (unsigned short*)dst, n);
    };
    cvt(x,     x_bf,   S_LEN * DIM_E);
    cvt(qr,    qr_bf,  S_LEN * QLORA_R);
    cvt(wq_b,  wqb_bf, NHEAD * DHEAD * QLORA_R);
    cvt(wk,    wk_bf,  DHEAD * DIM_E);
    cvt(w_wts, ww_bf,  NHEAD * DIM_E);

    // 2) GEMMs (B stored as N x K row-major == implicit transpose)
    {   // q = qr (S x QLORA) * wq_b^T (QLORA x H*DH): 32x64 per wave
        dim3 grid((DIM_E / 64 + 7) / 8, S_LEN / 32);
        gemm_bf16_wmma<2, 4><<<grid, 256, 0, stream>>>(qr_bf, wqb_bf, q_f32,
                                                       S_LEN, NHEAD * DHEAD, QLORA_R, 1.0f);
    }
    {   // k = x (S x DIM) * wk^T (DIM x DH): N=128 -> 2 n-tiles of 64
        dim3 grid(1, S_LEN / 32);
        gemm_bf16_wmma<2, 4><<<grid, 256, 0, stream>>>(x_bf, wk_bf, k_f32,
                                                       S_LEN, DHEAD, DIM_E, 1.0f);
    }
    {   // wgtT = w_weights (H x DIM) * x^T (DIM x S) * H^-0.5  -> h-major output
        dim3 grid((S_LEN / 64 + 7) / 8, NHEAD / 32);
        gemm_bf16_wmma<2, 4><<<grid, 256, 0, stream>>>(ww_bf, x_bf, wgtT,
                                                       NHEAD, S_LEN, DIM_E, 0.17677669529663687f);
    }

    // 3) post-processing (Hadamard dropped: Hm symmetric orthogonal, cancels in q.k^T)
    kpost_ln_rope<<<S_LEN / 8, 256, 0, stream>>>(k_f32, k_norm_w, k_norm_b, cosT, sinT, k_bf);
    qpost_rope<<<S_LEN * NHEAD / 8, 256, 0, stream>>>(q_f32, cosT, sinT, q_bf);

    // 4) fused scores + relu-weighted head reduce + causal mask
    {
        dim3 grid(S_LEN / 32, S_LEN / 128);
        score_wmma<<<grid, 256, 0, stream>>>(q_bf, k_bf, wgtT, isc);
    }

    // 5) top-512 indices per row
    topk_bitonic<<<S_LEN, 256, 0, stream>>>(isc, (int*)d_out);
}
